// MultiHeadAttentionLayer_48704929136766
// MI455X (gfx1250) — matrix operand order repaired and checked by reading, not compiled, not run on hardware
//
#include <hip/hip_runtime.h>
#include <hip/hip_bf16.h>
#include <cstdint>
#include <cstddef>

// ---------------------------------------------------------------------------
// MultiHeadAttention for MI455X (gfx1250), wave32, WMMA f16 16x16x32.
// B=8, N=2048, C_in=256, PROJ=256, H=4, DH=64. Scale = 1/sqrt(256) = 1/16.
// ---------------------------------------------------------------------------

typedef _Float16 v16h __attribute__((ext_vector_type(16)));
typedef float    v8f  __attribute__((ext_vector_type(8)));

#define NTOK   2048
#define NB     8
#define CIN    256
#define NHEAD  4
#define DHEAD  64
#define NROWS  (NB * NTOK)          // 16384
#define NP     (NROWS * CIN)        // 4194304 halfs per f16 tensor

static __device__ __forceinline__ v8f v8f_zero() {
    v8f z;
#pragma unroll
    for (int i = 0; i < 8; ++i) z[i] = 0.0f;
    return z;
}

static __device__ __forceinline__ v8f wmma_f16(v16h a, v16h b, v8f c) {
    return __builtin_amdgcn_wmma_f32_16x16x32_f16(false, a, false, b,
                                                  (short)0, c, false, false);
}

// ---------------------------------------------------------------------------
// Kernel 0: convert + transpose weights to f16.  WT[p][k] = W[k][p].
// ---------------------------------------------------------------------------
__global__ void mha_convw_kernel(const float* __restrict__ Wq,
                                 const float* __restrict__ Wk,
                                 const float* __restrict__ Wv,
                                 const float* __restrict__ Wo,
                                 _Float16* __restrict__ WqT,
                                 _Float16* __restrict__ WkT,
                                 _Float16* __restrict__ WvT,
                                 _Float16* __restrict__ WoT) {
    int t = blockIdx.x * blockDim.x + threadIdx.x;
    if (t < 65536) {
        int p = t >> 8, k = t & 255;
        WqT[t] = (_Float16)Wq[k * 256 + p];
    } else if (t < 131072) {
        int u = t - 65536; int p = u >> 8, k = u & 255;
        WkT[u] = (_Float16)Wk[k * 256 + p];
    } else if (t < 196608) {
        int u = t - 131072; int p = u >> 8, k = u & 255;
        WvT[u] = (_Float16)Wv[k * 256 + p];
    } else if (t < 212992) {
        int u = t - 196608; int p = u >> 8, k = u & 255;   // p in [0,64), k in [0,256)
        WoT[u] = (_Float16)Wo[k * 64 + p];
    }
}

// ---------------------------------------------------------------------------
// Kernel 1: QKV projection.  One wave computes one 16x16 output tile.
//   which==0: Q -> qh[b,n,p] (pre-scaled by 1/16)
//   which==1: K -> kh[b,n,p]
//   which==2: V -> vt[b,h,c,n]  (transposed per head for PV fragments)
// ---------------------------------------------------------------------------
__global__ __launch_bounds__(32) void mha_proj_kernel(
    const float* __restrict__ Xq, const float* __restrict__ Xk,
    const float* __restrict__ Xv,
    const _Float16* __restrict__ WqT, const _Float16* __restrict__ WkT,
    const _Float16* __restrict__ WvT,
    const float* __restrict__ bq, const float* __restrict__ bk,
    const float* __restrict__ bv,
    _Float16* __restrict__ qh, _Float16* __restrict__ kh,
    _Float16* __restrict__ vt) {
    const int lane = threadIdx.x;
    const int r  = lane & 15;
    const int hi = lane >> 4;
    const int which = blockIdx.z;

    const float*    X    = (which == 0) ? Xq  : (which == 1) ? Xk  : Xv;
    const _Float16* WT   = (which == 0) ? WqT : (which == 1) ? WkT : WvT;
    const float*    bias = (which == 0) ? bq  : (which == 1) ? bk  : bv;

    const long row0 = (long)blockIdx.x * 16;   // global row in [0,16384)
    const int  p0   = blockIdx.y * 16;         // output col tile

    v8f acc = v8f_zero();
#pragma unroll
    for (int k0 = 0; k0 < 256; k0 += 32) {
        // A fragment: row = row0 + r; elements k = k0 + 16hi + i
        const float* ap = X + (row0 + r) * 256 + k0 + 16 * hi;
        v16h a;
#pragma unroll
        for (int i = 0; i < 8; ++i) {
            a[i]     = (_Float16)ap[i];
            a[8 + i] = (_Float16)ap[8 + i];
        }
        // B fragment: col = p0 + r; elements k = k0 + 16hi + i  (WT contiguous)
        const _Float16* bp = WT + (size_t)(p0 + r) * 256 + k0 + 16 * hi;
        v16h bf;
#pragma unroll
        for (int i = 0; i < 16; ++i) bf[i] = bp[i];
        acc = wmma_f16(a, bf, acc);
    }

    const float biasv = bias[p0 + r];
    const float scale = (which == 0) ? 0.0625f : 1.0f;   // 1/sqrt(256) baked into Q
#pragma unroll
    for (int j = 0; j < 8; ++j) {
        float val = (acc[j] + biasv) * scale;
        long n = row0 + j + 8 * hi;        // global row
        int  p = p0 + r;                   // global proj col
        if (which == 2) {
            long bb = n >> 11;             // batch
            long nn = n & 2047;            // token
            vt[(((bb * NHEAD) + (p >> 6)) * 64 + (p & 63)) * (long)NTOK + nn] =
                (_Float16)val;
        } else {
            _Float16* dst = (which == 0) ? qh : kh;
            dst[n * 256 + p] = (_Float16)val;
        }
    }
}

// ---------------------------------------------------------------------------
// Kernel 2: flash attention.  One wave per (b, h, 16-row q tile).
// ---------------------------------------------------------------------------
#define PSTR 40   // LDS row stride (halfs) to avoid bank conflicts

__global__ __launch_bounds__(32) void mha_attn_kernel(
    const _Float16* __restrict__ qh, const _Float16* __restrict__ kh,
    const _Float16* __restrict__ vt, _Float16* __restrict__ attbuf) {
    __shared__ _Float16 P[16 * PSTR];

    const int lane = threadIdx.x;
    const int r  = lane & 15;
    const int hi = lane >> 4;
    const int qt = blockIdx.x;      // q row tile [0,128)
    const int h  = blockIdx.y;      // head
    const int b  = blockIdx.z;      // batch

    const long rowbase = (long)b * NTOK + qt * 16;   // global q row base

    // Q A-fragments for channels [0,32) and [32,64); stay resident.
    v16h qa0, qa1;
    {
        const _Float16* ap = qh + (rowbase + r) * 256 + h * 64 + 16 * hi;
#pragma unroll
        for (int i = 0; i < 8; ++i) {
            qa0[i]     = ap[i];
            qa0[8 + i] = ap[8 + i];
            qa1[i]     = ap[32 + i];
            qa1[8 + i] = ap[40 + i];
        }
    }

    v8f o[4];
#pragma unroll
    for (int c = 0; c < 4; ++c) o[c] = v8f_zero();
    float mvec[8], lvec[8];
#pragma unroll
    for (int j = 0; j < 8; ++j) { mvec[j] = -1.0e30f; lvec[j] = 0.0f; }

    for (int kv = 0; kv < NTOK; kv += 32) {
        // ---- scores: s0 = Q @ K^T[:, kv..kv+15], s1 = cols kv+16..kv+31 ----
        v8f s0 = v8f_zero(), s1 = v8f_zero();
        {
            const _Float16* bp =
                kh + ((long)b * NTOK + kv + r) * 256 + h * 64 + 16 * hi;
            v16h kb;
#pragma unroll
            for (int i = 0; i < 16; ++i) kb[i] = bp[i];
            s0 = wmma_f16(qa0, kb, s0);
#pragma unroll
            for (int i = 0; i < 16; ++i) kb[i] = bp[32 + i];
            s0 = wmma_f16(qa1, kb, s0);
        }
        {
            const _Float16* bp =
                kh + ((long)b * NTOK + kv + 16 + r) * 256 + h * 64 + 16 * hi;
            v16h kb;
#pragma unroll
            for (int i = 0; i < 16; ++i) kb[i] = bp[i];
            s1 = wmma_f16(qa0, kb, s1);
#pragma unroll
            for (int i = 0; i < 16; ++i) kb[i] = bp[32 + i];
            s1 = wmma_f16(qa1, kb, s1);
        }

        // ---- online softmax over this 32-key block ----
        // C/D layout: VGPR j holds row (j + 8*hi), lane%16 is the column.
#pragma unroll
        for (int j = 0; j < 8; ++j) {
            float mp = fmaxf(s0[j], s1[j]);
            mp = fmaxf(mp, __shfl_xor(mp, 1, 32));
            mp = fmaxf(mp, __shfl_xor(mp, 2, 32));
            mp = fmaxf(mp, __shfl_xor(mp, 4, 32));
            mp = fmaxf(mp, __shfl_xor(mp, 8, 32));
            float mnew  = fmaxf(mvec[j], mp);
            float alpha = __expf(mvec[j] - mnew);
            mvec[j] = mnew;
            float p0v = __expf(s0[j] - mnew);
            float p1v = __expf(s1[j] - mnew);
            float ps  = p0v + p1v;
            ps += __shfl_xor(ps, 1, 32);
            ps += __shfl_xor(ps, 2, 32);
            ps += __shfl_xor(ps, 4, 32);
            ps += __shfl_xor(ps, 8, 32);
            lvec[j] = lvec[j] * alpha + ps;
#pragma unroll
            for (int c = 0; c < 4; ++c) o[c][j] *= alpha;
            // Stage P (16x32) into LDS in row-major halfs.
            P[(j + 8 * hi) * PSTR + r]      = (_Float16)p0v;
            P[(j + 8 * hi) * PSTR + 16 + r] = (_Float16)p1v;
        }
        __syncthreads();

        // ---- reload P as an A fragment (row = lane%16) ----
        v16h pa;
        {
            const _Float16* pp = &P[r * PSTR + 16 * hi];
#pragma unroll
            for (int i = 0; i < 8; ++i) {
                pa[i]     = pp[i];
                pa[8 + i] = pp[8 + i];
            }
        }
        __syncthreads();

        // ---- O += P @ V (vt layout gives contiguous B fragments) ----
#pragma unroll
        for (int c = 0; c < 4; ++c) {
            const _Float16* vp =
                vt + ((long)(b * NHEAD + h) * 64 + c * 16 + r) * (long)NTOK +
                kv + 16 * hi;
            v16h vb;
#pragma unroll
            for (int i = 0; i < 16; ++i) vb[i] = vp[i];
            o[c] = wmma_f16(pa, vb, o[c]);
        }
    }

    // ---- normalize and emit f16 into attbuf[b,n, h*64 + c] ----
#pragma unroll
    for (int c = 0; c < 4; ++c) {
#pragma unroll
        for (int j = 0; j < 8; ++j) {
            float val = o[c][j] / lvec[j];
            long n = rowbase + j + 8 * hi;
            attbuf[n * 256 + h * 64 + c * 16 + r] = (_Float16)val;
        }
    }
}

// ---------------------------------------------------------------------------
// Kernel 3: output projection  attbuf(16384x256) @ WoT^T(256x64) + bo -> f32.
// ---------------------------------------------------------------------------
__global__ __launch_bounds__(32) void mha_outproj_kernel(
    const _Float16* __restrict__ attbuf, const _Float16* __restrict__ WoT,
    const float* __restrict__ bo, float* __restrict__ out) {
    const int lane = threadIdx.x;
    const int r  = lane & 15;
    const int hi = lane >> 4;
    const long row0 = (long)blockIdx.x * 16;
    const int  c0   = blockIdx.y * 16;   // output col tile [0,64)

    v8f acc = v8f_zero();
#pragma unroll
    for (int k0 = 0; k0 < 256; k0 += 32) {
        const _Float16* ap = attbuf + (row0 + r) * 256 + k0 + 16 * hi;
        v16h a;
#pragma unroll
        for (int i = 0; i < 8; ++i) {
            a[i]     = ap[i];
            a[8 + i] = ap[8 + i];
        }
        const _Float16* bp = WoT + (size_t)(c0 + r) * 256 + k0 + 16 * hi;
        v16h bf;
#pragma unroll
        for (int i = 0; i < 16; ++i) bf[i] = bp[i];
        acc = wmma_f16(a, bf, acc);
    }
    const float biasv = bo[c0 + r];
#pragma unroll
    for (int j = 0; j < 8; ++j) {
        long n = row0 + j + 8 * hi;
        out[n * 64 + c0 + r] = acc[j] + biasv;
    }
}

// ---------------------------------------------------------------------------
// Host-side launch.
// d_in order: query, key, value, Wq, bq, Wk, bk, Wv, bv, Wo, bo
// ---------------------------------------------------------------------------
extern "C" void kernel_launch(void* const* d_in, const int* in_sizes, int n_in,
                              void* d_out, int out_size, void* d_ws,
                              size_t ws_size, hipStream_t stream) {
    (void)in_sizes; (void)n_in; (void)out_size; (void)ws_size;

    const float* query = (const float*)d_in[0];
    const float* key_  = (const float*)d_in[1];
    const float* value = (const float*)d_in[2];
    const float* Wq    = (const float*)d_in[3];
    const float* bq    = (const float*)d_in[4];
    const float* Wk    = (const float*)d_in[5];
    const float* bk    = (const float*)d_in[6];
    const float* Wv    = (const float*)d_in[7];
    const float* bv    = (const float*)d_in[8];
    const float* Wo    = (const float*)d_in[9];
    const float* bo    = (const float*)d_in[10];
    float* out = (float*)d_out;

    // Workspace layout (halfs): qh | kh | vt | attbuf | WqT | WkT | WvT | WoT
    _Float16* w      = (_Float16*)d_ws;
    _Float16* qh     = w;
    _Float16* kh     = w + (size_t)NP;
    _Float16* vt     = w + (size_t)2 * NP;
    _Float16* attbuf = w + (size_t)3 * NP;
    _Float16* WqT    = w + (size_t)4 * NP;
    _Float16* WkT    = WqT + 65536;
    _Float16* WvT    = WkT + 65536;
    _Float16* WoT    = WvT + 65536;

    // 0) weights -> f16 transposed
    mha_convw_kernel<<<832, 256, 0, stream>>>(Wq, Wk, Wv, Wo,
                                              WqT, WkT, WvT, WoT);
    // 1) QKV projections (z = which)
    mha_proj_kernel<<<dim3(NROWS / 16, 16, 3), 32, 0, stream>>>(
        query, key_, value, WqT, WkT, WvT, bq, bk, bv, qh, kh, vt);
    // 2) flash attention
    mha_attn_kernel<<<dim3(NTOK / 16, NHEAD, NB), 32, 0, stream>>>(
        qh, kh, vt, attbuf);
    // 3) output projection
    mha_outproj_kernel<<<dim3(NROWS / 16, DHEAD / 16), 32, 0, stream>>>(
        attbuf, WoT, bo, out);
}
